// SelectiveSSM_36756330119909
// MI455X (gfx1250) — matrix operand order repaired and checked by reading, not compiled, not run on hardware
//
#include <hip/hip_runtime.h>
#include <hip/hip_bf16.h>
#include <stdint.h>

// ---------------------------------------------------------------------------
// Selective SSM (Mamba-style) for MI455X / gfx1250, wave32 + WMMA bf16.
//   0) fp32 -> bf16 converts of x, W_in, W_out, W_dt[:16]  (streaming)
//   1) xz = x @ W_in^T            GEMM 4096x8192x2048      (v_wmma bf16)
//   2) depthwise conv4 + bias + SiLU -> xs (bf16)          (streaming)
//   3) dt = softplus(xs @ W_dt[:16]^T + b_dt[:16])         (v_wmma bf16, N=16)
//   4) sequential scan (1 wave, lanes = 2 batches x 16 states)
//   5) y = (res*D_INNER) * sigmoid(gate) -> bf16           (streaming)
//   6) out = y @ W_out^T          GEMM 4096x2048x4096      (v_wmma bf16, f32 out)
// GEMMs use async global->LDS (ASYNCcnt) double buffering.
// ---------------------------------------------------------------------------

typedef __attribute__((ext_vector_type(16))) __bf16 v16bf;
typedef __attribute__((ext_vector_type(8)))  float  v8f;
typedef int v4i __attribute__((vector_size(16)));   // b128 payload type

#define B_SZ 2
#define S_SZ 2048
#define DM   2048
#define DI   4096
#define DS   16
#define DC   4
#define ROWS (B_SZ * S_SZ)   // 4096 flattened (b,s) rows

#define GLOBAL_AS __attribute__((address_space(1)))
#define LDS_AS    __attribute__((address_space(3)))

// ---- bf16 helpers ---------------------------------------------------------
__device__ __forceinline__ unsigned short f2bf(float f) {
  unsigned int u = __float_as_uint(f);
  unsigned int r = u + 0x7FFFu + ((u >> 16) & 1u);   // round-to-nearest-even
  return (unsigned short)(r >> 16);
}
__device__ __forceinline__ float bf2f(unsigned short h) {
  return __uint_as_float(((unsigned int)h) << 16);
}
#if __has_builtin(__builtin_amdgcn_cvt_pk_bf16_f32)
typedef __attribute__((ext_vector_type(2))) __bf16 v2bf;
__device__ __forceinline__ unsigned int pk2(float a, float b) {
  union { v2bf v; unsigned int u; } cv;
  cv.v = __builtin_amdgcn_cvt_pk_bf16_f32(a, b);
  return cv.u;
}
#else
__device__ __forceinline__ unsigned int pk2(float a, float b) {
  return (unsigned int)f2bf(a) | ((unsigned int)f2bf(b) << 16);
}
#endif
__device__ __forceinline__ float lo16(unsigned int w) { return __uint_as_float(w << 16); }
__device__ __forceinline__ float hi16(unsigned int w) { return __uint_as_float(w & 0xFFFF0000u); }

union FragBF { v16bf v; uint4 q[2]; };

// ---- async global -> LDS (CDNA5, ASYNCcnt) --------------------------------
#if __has_builtin(__builtin_amdgcn_global_load_async_to_lds_b128)
#define USE_ASYNC_LDS 1
__device__ __forceinline__ void async_ld_b128(const unsigned short* g, unsigned short* l) {
  __builtin_amdgcn_global_load_async_to_lds_b128(
      (GLOBAL_AS v4i*)g, (LDS_AS v4i*)l, 0, 0);
}
#else
#define USE_ASYNC_LDS 0
#endif

__device__ __forceinline__ void wait_async0() {
#if __has_builtin(__builtin_amdgcn_s_wait_asynccnt)
  __builtin_amdgcn_s_wait_asynccnt(0);
#else
  asm volatile("s_wait_asynccnt 0" ::: "memory");
#endif
}

// ---- 0) fp32 -> bf16 bulk convert (4 elems/thread) ------------------------
__global__ void cvt_f32_to_bf16_kernel(const float* __restrict__ in,
                                       unsigned short* __restrict__ out, long n4) {
  long t = (long)blockIdx.x * blockDim.x + threadIdx.x;
  if (t >= n4) return;
  const float4 f = *reinterpret_cast<const float4*>(in + t * 4);
  uint2 o; o.x = pk2(f.x, f.y); o.y = pk2(f.z, f.w);
  *reinterpret_cast<uint2*>(out + t * 4) = o;
}

// ---- 1/6) NT GEMM, bf16 in / f32 accumulate, WMMA 16x16x32 ----------------
// C[M,N] = A[M,K] * B[N,K]^T.  Block tile 128x128x32, 8 waves (4 in M, 2 in N),
// each wave: 2x4 tiles of 16x16.  LDS rows padded to 40 bf16 (80B) for banks.
// Async path: double-buffered LDS, tile k+1 loads overlap tile k WMMAs.
template <bool STORE_BF16>
__global__ __launch_bounds__(256) void gemm_nt_bf16_wmma(
    const unsigned short* __restrict__ A,
    const unsigned short* __restrict__ Bm,
    float* __restrict__ Cf,
    unsigned short* __restrict__ Cb,
    int M, int N, int K) {
#if USE_ASYNC_LDS
  __shared__ unsigned short sA[2][128 * 40];
  __shared__ unsigned short sB[2][128 * 40];
#else
  __shared__ unsigned short sA[1][128 * 40];
  __shared__ unsigned short sB[1][128 * 40];
#endif

  const int tid  = threadIdx.x;
  const int lane = tid & 31;
  const int wave = tid >> 5;
  const int wm   = wave & 3;      // wave row (M)
  const int wn   = wave >> 2;     // wave col (N)
  const int lrow = lane & 15;
  const int lhi  = lane >> 4;     // lane half

  const int ldRow = tid >> 2;           // 0..63, 2 passes cover 128 rows
  const int ldCol = (tid & 3) * 8;      // 8 bf16 (16B) per thread per row

  const size_t aBase = (size_t)blockIdx.y * 128 * (size_t)K;
  const size_t bBase = (size_t)blockIdx.x * 128 * (size_t)K;

  v8f acc[2][4] = {};

#if USE_ASYNC_LDS
  // prologue: stage tile 0 into buffer 0
  {
    const int kt = 0;
#pragma unroll
    for (int p = 0; p < 2; ++p) {
      const int r = p * 64 + ldRow;
      async_ld_b128(A  + aBase + (size_t)r * K + kt + ldCol, &sA[0][r * 40 + ldCol]);
      async_ld_b128(Bm + bBase + (size_t)r * K + kt + ldCol, &sB[0][r * 40 + ldCol]);
    }
  }
  for (int kt = 0, it = 0; kt < K; kt += 32, ++it) {
    const int buf = it & 1;
    wait_async0();        // my tile-kt async loads complete
    __syncthreads();      // everyone's tile kt visible; prev buffer free
    if (kt + 32 < K) {    // overlap: stage tile kt+32 into the other buffer
      const int nb = buf ^ 1;
#pragma unroll
      for (int p = 0; p < 2; ++p) {
        const int r = p * 64 + ldRow;
        async_ld_b128(A  + aBase + (size_t)r * K + kt + 32 + ldCol, &sA[nb][r * 40 + ldCol]);
        async_ld_b128(Bm + bBase + (size_t)r * K + kt + 32 + ldCol, &sB[nb][r * 40 + ldCol]);
      }
    }
#else
  for (int kt = 0; kt < K; kt += 32) {
    const int buf = 0;
#pragma unroll
    for (int p = 0; p < 2; ++p) {
      const int r = p * 64 + ldRow;
      const uint4 va = *reinterpret_cast<const uint4*>(A  + aBase + (size_t)r * K + kt + ldCol);
      const uint4 vb = *reinterpret_cast<const uint4*>(Bm + bBase + (size_t)r * K + kt + ldCol);
      *reinterpret_cast<uint4*>(&sA[0][r * 40 + ldCol]) = va;
      *reinterpret_cast<uint4*>(&sB[0][r * 40 + ldCol]) = vb;
    }
    if (kt + 32 < K) {
      __builtin_prefetch(A  + aBase + (size_t)ldRow * K + kt + 32, 0, 1);
      __builtin_prefetch(Bm + bBase + (size_t)ldRow * K + kt + 32, 0, 1);
    }
    __syncthreads();
#endif

    FragBF af[2], bf4[4];
#pragma unroll
    for (int i = 0; i < 2; ++i) {   // A 16x32 layout: per-lane 8-elem K chunks
      const int r = wm * 32 + i * 16 + lrow;
      af[i].q[0] = *reinterpret_cast<const uint4*>(&sA[buf][r * 40 + 8 * lhi]);
      af[i].q[1] = *reinterpret_cast<const uint4*>(&sA[buf][r * 40 + 16 + 8 * lhi]);
    }
#pragma unroll
    for (int j = 0; j < 4; ++j) {   // B 32x16 layout: per-lane 8-elem K chunks
      const int r = wn * 64 + j * 16 + lrow;
      bf4[j].q[0] = *reinterpret_cast<const uint4*>(&sB[buf][r * 40 + 8 * lhi]);
      bf4[j].q[1] = *reinterpret_cast<const uint4*>(&sB[buf][r * 40 + 16 + 8 * lhi]);
    }
#pragma unroll
    for (int i = 0; i < 2; ++i)
#pragma unroll
      for (int j = 0; j < 4; ++j)
        acc[i][j] = __builtin_amdgcn_wmma_f32_16x16x32_bf16(
            false, af[i].v, false, bf4[j].v, (short)0, acc[i][j], false, false);
#if !USE_ASYNC_LDS
    __syncthreads();
#endif
  }

  // C layout: VGPR v -> M = v + 8*laneHalf, N = lane&15
#pragma unroll
  for (int i = 0; i < 2; ++i)
#pragma unroll
    for (int j = 0; j < 4; ++j) {
      const int baseM = blockIdx.y * 128 + wm * 32 + i * 16 + 8 * lhi;
      const int baseN = blockIdx.x * 128 + wn * 64 + j * 16 + lrow;
#pragma unroll
      for (int v = 0; v < 8; ++v) {
        const size_t off = (size_t)(baseM + v) * N + baseN;
        if (STORE_BF16) Cb[off] = f2bf(acc[i][j][v]);
        else            Cf[off] = acc[i][j][v];
      }
    }
}

// ---- 2) depthwise causal conv (window 4) + bias + SiLU --------------------
__global__ void conv_silu_kernel(const unsigned short* __restrict__ xz,
                                 const float* __restrict__ conv_w,
                                 const float* __restrict__ conv_b,
                                 unsigned short* __restrict__ xs) {
  const int t = blockIdx.x * blockDim.x + threadIdx.x;  // 8 channels / thread
  if (t >= ROWS * (DI / 8)) return;
  const int c0 = (t & (DI / 8 - 1)) * 8;
  const int r  = t >> 9;            // DI/8 == 512
  const int s  = r & (S_SZ - 1);
  float acc[8];
#pragma unroll
  for (int u = 0; u < 8; ++u) acc[u] = conv_b[c0 + u];
#pragma unroll
  for (int j = 0; j < DC; ++j) {
    const int sp = s - (DC - 1) + j;
    if (sp < 0) continue;
    const uint4 q = *reinterpret_cast<const uint4*>(xz + (size_t)(r - (DC - 1) + j) * (2 * DI) + c0);
    const float xv[8] = { lo16(q.x), hi16(q.x), lo16(q.y), hi16(q.y),
                          lo16(q.z), hi16(q.z), lo16(q.w), hi16(q.w) };
#pragma unroll
    for (int u = 0; u < 8; ++u) acc[u] += xv[u] * conv_w[(c0 + u) * DC + j];
  }
  float sv[8];
#pragma unroll
  for (int u = 0; u < 8; ++u) { const float a = acc[u]; sv[u] = a / (1.0f + __expf(-a)); }
  uint4 o;
  o.x = pk2(sv[0], sv[1]); o.y = pk2(sv[2], sv[3]);
  o.z = pk2(sv[4], sv[5]); o.w = pk2(sv[6], sv[7]);
  *reinterpret_cast<uint4*>(xs + (size_t)r * DI + c0) = o;
}

// ---- 3) dt = softplus(xs @ W_dt[:16]^T + b_dt[:16]), one wave / 16 rows ---
// Both fragments now load pre-converted bf16: loop body is pure loads + WMMA.
__global__ void dt_wmma_kernel(const unsigned short* __restrict__ xs,
                               const unsigned short* __restrict__ Wdtbf,  // [16][DI] bf16
                               const float* __restrict__ b_dt,
                               float* __restrict__ dt) {
  const int lane = threadIdx.x & 31;
  const int lrow = lane & 15;
  const int lhi  = lane >> 4;
  const int m0   = blockIdx.x * 16;
  v8f acc = {};
  const size_t arow = (size_t)(m0 + lrow) * DI;
  const size_t brow = (size_t)lrow * DI;
  for (int k = 0; k < DI; k += 32) {
    FragBF a, b;
    a.q[0] = *reinterpret_cast<const uint4*>(xs + arow + k + 8 * lhi);
    a.q[1] = *reinterpret_cast<const uint4*>(xs + arow + k + 16 + 8 * lhi);
    b.q[0] = *reinterpret_cast<const uint4*>(Wdtbf + brow + k + 8 * lhi);
    b.q[1] = *reinterpret_cast<const uint4*>(Wdtbf + brow + k + 16 + 8 * lhi);
    acc = __builtin_amdgcn_wmma_f32_16x16x32_bf16(false, a.v, false, b.v,
                                                  (short)0, acc, false, false);
  }
  const int n = lrow;
  const float bd = b_dt[n];
#pragma unroll
  for (int v = 0; v < 8; ++v) {
    const int r = m0 + v + 8 * lhi;
    const float x = acc[v] + bd;
    // stable softplus: max(x,0) + log(1 + exp(-|x|))  (1 exp + 1 log)
    const float sp = fmaxf(x, 0.0f) + __logf(1.0f + __expf(-fabsf(x)));
    dt[(size_t)r * DS + n] = sp;
  }
}

// ---- 4) sequential selective scan: lanes 0-15 = b0 states, 16-31 = b1 -----
__global__ void scan_kernel(const float* __restrict__ dt,
                            const unsigned short* __restrict__ xs,
                            const float* __restrict__ A_log,
                            float* __restrict__ res) {
  const int lane = threadIdx.x;
  const int n = lane & 15;
  const int b = lane >> 4;
  const float Ac = -__expf(A_log[n]);
  float h = 0.0f;
  for (int s = 0; s < S_SZ; ++s) {
    const int r = b * S_SZ + s;
    const float dtv = dt[(size_t)r * DS + n];
    const float uv  = bf2f(xs[(size_t)r * DI + n]);
    h = h * (1.0f + dtv) + uv;
    float val = h * Ac;
    val += __shfl_xor(val, 1, 32);
    val += __shfl_xor(val, 2, 32);
    val += __shfl_xor(val, 4, 32);
    val += __shfl_xor(val, 8, 32);   // sum over 16 states within each half
    if (n == 0) res[r] = val;
  }
}

// ---- 5) y = (res * D_INNER) * sigmoid(gate) -------------------------------
__global__ void gate_mul_kernel(const unsigned short* __restrict__ xz,
                                const float* __restrict__ res,
                                unsigned short* __restrict__ y) {
  const int t = blockIdx.x * blockDim.x + threadIdx.x;
  if (t >= ROWS * (DI / 8)) return;
  const int e0 = (t & (DI / 8 - 1)) * 8;
  const int r  = t >> 9;
  const float scale = res[r] * (float)DI;
  const uint4 g = *reinterpret_cast<const uint4*>(xz + (size_t)r * (2 * DI) + DI + e0);
  const float gv[8] = { lo16(g.x), hi16(g.x), lo16(g.y), hi16(g.y),
                        lo16(g.z), hi16(g.z), lo16(g.w), hi16(g.w) };
  float yv[8];
#pragma unroll
  for (int u = 0; u < 8; ++u) yv[u] = scale / (1.0f + __expf(-gv[u]));
  uint4 o;
  o.x = pk2(yv[0], yv[1]); o.y = pk2(yv[2], yv[3]);
  o.z = pk2(yv[4], yv[5]); o.w = pk2(yv[6], yv[7]);
  *reinterpret_cast<uint4*>(y + (size_t)r * DI + e0) = o;
}

// ---------------------------------------------------------------------------
extern "C" void kernel_launch(void* const* d_in, const int* in_sizes, int n_in,
                              void* d_out, int out_size, void* d_ws, size_t ws_size,
                              hipStream_t stream) {
  (void)in_sizes; (void)n_in; (void)out_size; (void)ws_size;
  const float* x      = (const float*)d_in[0];
  const float* W_in   = (const float*)d_in[1];
  const float* conv_w = (const float*)d_in[2];
  const float* conv_b = (const float*)d_in[3];
  const float* W_dt   = (const float*)d_in[4];
  const float* b_dt   = (const float*)d_in[5];
  const float* A_log  = (const float*)d_in[6];
  const float* W_out  = (const float*)d_in[7];
  float* out = (float*)d_out;

  // Workspace layout (~160.5 MB total):
  unsigned short* xbf   = (unsigned short*)d_ws;            //  16 MB x bf16
  unsigned short* wibf  = xbf  + (size_t)ROWS * DM;         //  32 MB W_in bf16
  unsigned short* wobf  = wibf + (size_t)(2 * DI) * DM;     //  16 MB W_out bf16
  unsigned short* xz    = wobf + (size_t)DM * DI;           //  64 MB xz bf16
  unsigned short* xs    = xz   + (size_t)ROWS * (2 * DI);   //  32 MB xs bf16
  float*          dt    = (float*)(xs + (size_t)ROWS * DI); // 256 KB dt f32
  float*          res   = dt + (size_t)ROWS * DS;           //  16 KB res f32
  unsigned short* wdtbf = (unsigned short*)(res + ROWS);    // 128 KB W_dt[:16] bf16
  unsigned short* ybf   = xs;  // y reuses xs (scan finished before gate_mul)

  const int TPB = 256;
  cvt_f32_to_bf16_kernel<<<(ROWS * DM / 4 + TPB - 1) / TPB, TPB, 0, stream>>>(x, xbf, ROWS * DM / 4);
  cvt_f32_to_bf16_kernel<<<((2 * DI) * DM / 4 + TPB - 1) / TPB, TPB, 0, stream>>>(W_in, wibf, (2 * DI) * DM / 4);
  cvt_f32_to_bf16_kernel<<<(DM * DI / 4 + TPB - 1) / TPB, TPB, 0, stream>>>(W_out, wobf, DM * DI / 4);
  cvt_f32_to_bf16_kernel<<<(DS * DI / 4 + TPB - 1) / TPB, TPB, 0, stream>>>(W_dt, wdtbf, DS * DI / 4);

  // xz = x @ W_in^T : M=4096, N=8192, K=2048 (bf16 out)
  gemm_nt_bf16_wmma<true><<<dim3((2 * DI) / 128, ROWS / 128), 256, 0, stream>>>(
      xbf, wibf, nullptr, xz, ROWS, 2 * DI, DM);

  conv_silu_kernel<<<(ROWS * (DI / 8) + TPB - 1) / TPB, TPB, 0, stream>>>(xz, conv_w, conv_b, xs);

  dt_wmma_kernel<<<ROWS / 16, 32, 0, stream>>>(xs, wdtbf, b_dt, dt);

  scan_kernel<<<1, 32, 0, stream>>>(dt, xs, A_log, res);

  gate_mul_kernel<<<(ROWS * (DI / 8) + TPB - 1) / TPB, TPB, 0, stream>>>(xz, res, ybf);

  // out = y @ W_out^T : M=4096, N=2048, K=4096 (f32 out)
  gemm_nt_bf16_wmma<false><<<dim3(DM / 128, ROWS / 128), 256, 0, stream>>>(
      ybf, wobf, out, nullptr, ROWS, DM, DI);
}
